// SoapBPNN_75771813036181
// MI455X (gfx1250) — matrix operand order repaired and checked by reading, not compiled, not run on hardware
//
#include <hip/hip_runtime.h>
#include <stdint.h>

// ---------------- types / helpers ----------------
typedef __bf16 bf16_t;
typedef bf16_t v16bf __attribute__((ext_vector_type(16)));
typedef float  v8f   __attribute__((ext_vector_type(8)));

__device__ __forceinline__ uint16_t f2bf(float x) {
    uint32_t b = __float_as_uint(x);
    uint32_t r = b + 0x7FFFu + ((b >> 16) & 1u);   // round-to-nearest-even
    return (uint16_t)(r >> 16);
}
__device__ __forceinline__ float bf2f(uint16_t h) {
    return __uint_as_float(((uint32_t)h) << 16);
}

#define BM 64
#define BN 256
#define BK 32
#define SAS (BK + 8)    // sA row stride (elements): +16B pad -> conflict-free b128 frag reads
#define SBS (BK + 8)    // sB row stride
#define HTS (BN + 8)    // hT row stride

// ---------------- weight convert + transpose: fp32 [S][K][256] -> bf16 [S][256][Kpad]
__global__ __launch_bounds__(256) void convert_weight_T(
    const float* __restrict__ W, uint16_t* __restrict__ Wt, int K, int Kpad)
{
    const int n = blockIdx.x;            // 0..255 output column
    const int s = blockIdx.y;            // species
    const float*   Ws  = W  + (size_t)s * K * 256;
    uint16_t*      Wts = Wt + (size_t)s * 256 * Kpad;
    for (int k = threadIdx.x; k < Kpad; k += blockDim.x) {
        float v = (k < K) ? Ws[(size_t)k * 256 + n] : 0.0f;
        Wts[(size_t)n * Kpad + k] = f2bf(v);
    }
}

// ---------------- zero output ----------------
__global__ void zero_out(float* __restrict__ out, int n) {
    for (int i = threadIdx.x + blockIdx.x * blockDim.x; i < n; i += blockDim.x * gridDim.x)
        out[i] = 0.0f;
}

// ---------------- fully fused MLP: 3 GEMM layers + output dot + scatter ----------------
// Block: 256 threads = 8 waves; tile = 64 rows x 256 cols (full hidden width), so each
// layer's output tile (in LDS) is the complete A input of the next layer -> hidden
// activations never touch HBM. Wave tile 32x64 = 2x4 WMMA accumulators.
__global__ __launch_bounds__(256) void fused_mlp(
    const float* __restrict__ feats,    // fp32 [S][Nrows][K0]
    const uint16_t* __restrict__ Wt0,   // bf16 [S][256][K0pad] (K-contiguous)
    const uint16_t* __restrict__ Wt1,   // bf16 [S][256][256]
    const uint16_t* __restrict__ Wt2,   // bf16 [S][256][256]
    const float* __restrict__ b0,       // [S][256]
    const float* __restrict__ b1,       // [S][256]
    const float* __restrict__ b2,       // [S][256]
    const float* __restrict__ Wout,     // [S][256]
    const float* __restrict__ bout,     // [S]
    const int* __restrict__ ids,        // [S][Nrows]
    float* __restrict__ out,            // [NUM_STRUCTURES]
    int Nrows, int K0, int K0pad)
{
    __shared__ uint16_t sA[2][BM * SAS];   // staged feats tile   2 x 5 KB
    __shared__ uint16_t sB[2][BN * SBS];   // staged weight tile  2 x 20 KB
    __shared__ uint16_t hT[BM * HTS];      // hidden tile (bf16)  33 KB

    const int s    = blockIdx.z;
    const int m0   = blockIdx.x * BM;
    const int tid  = threadIdx.x;
    const int lane = tid & 31;
    const int wave = tid >> 5;
    const int waveM = wave & 1;         // 2 M groups of 32
    const int waveN = wave >> 1;        // 4 N groups of 64
    const int half  = lane >> 4;        // K-half selector per ISA 16-bit layout
    const int l     = lane & 15;        // row/col within 16x16 tile

    const float*    Af   = feats + (size_t)s * Nrows * K0;
    const uint16_t* Wt0s = Wt0 + (size_t)s * BN * K0pad;
    const uint16_t* Wt1s = Wt1 + (size_t)s * BN * BN;
    const uint16_t* Wt2s = Wt2 + (size_t)s * BN * BN;

    // per-thread feats staging coords: 64 rows x 32 k, 8 fp32 -> 8 bf16 per thread
    const int  arow  = tid >> 2;          // 0..63
    const int  akg   = (tid & 3) * 8;     // 0,8,16,24
    const int  grow  = m0 + arow;
    const bool rowOK = grow < Nrows;

    auto loadA = [&](int kt) -> uint4 {   // K0 % 8 == 0: chunk fully in-range or fully pad
        uint4 q = make_uint4(0u, 0u, 0u, 0u);
        if (rowOK && (kt + akg + 8) <= K0) {
            const float4* p = (const float4*)(Af + (size_t)grow * K0 + kt + akg);
            float4 x0 = p[0], x1 = p[1];
            union { uint16_t u[8]; uint4 q; } t;
            t.u[0] = f2bf(x0.x); t.u[1] = f2bf(x0.y);
            t.u[2] = f2bf(x0.z); t.u[3] = f2bf(x0.w);
            t.u[4] = f2bf(x1.x); t.u[5] = f2bf(x1.y);
            t.u[6] = f2bf(x1.z); t.u[7] = f2bf(x1.w);
            q = t.q;
        }
        return q;
    };

    union Frag { v16bf v; uint4 q[2]; };
    v8f acc[2][4];

    auto zeroAcc = [&]() {
        #pragma unroll
        for (int mi = 0; mi < 2; ++mi)
            #pragma unroll
            for (int ni = 0; ni < 4; ++ni) { v8f z = {}; acc[mi][ni] = z; }
    };
    // bias + SiLU on acc, write bf16 tile into hT
    auto epilogueToLds = [&](const float* bl) {
        #pragma unroll
        for (int mi = 0; mi < 2; ++mi) {
            #pragma unroll
            for (int ni = 0; ni < 4; ++ni) {
                const int col = waveN * 64 + ni * 16 + l;
                const float bv = bl[col];
                #pragma unroll
                for (int r = 0; r < 8; ++r) {
                    const int row = waveM * 32 + mi * 16 + half * 8 + r;
                    float x = acc[mi][ni][r] + bv;
                    float y = x / (1.0f + __expf(-x));          // SiLU
                    hT[row * HTS + col] = f2bf(y);
                }
            }
        }
    };

    // ================= stage 1: h0 = silu(feats x W0 + b0) =================
    zeroAcc();
    {   // prologue: stage k-tile 0 into buffer 0
        uint4 aq = loadA(0);
        const uint4* bq = (const uint4*)(Wt0s + (size_t)tid * K0pad);
        uint4 q0 = bq[0], q1 = bq[1], q2 = bq[2], q3 = bq[3];
        *(uint4*)&sA[0][arow * SAS + akg] = aq;
        uint4* sq = (uint4*)&sB[0][tid * SBS];
        sq[0] = q0; sq[1] = q1; sq[2] = q2; sq[3] = q3;
    }
    int buf = 0;
    for (int kt = 0; kt < K0pad; kt += BK, buf ^= 1) {
        __syncthreads();   // staged stores to `buf` visible; prior reads of buf^1 done

        const bool hasNext = (kt + BK) < K0pad;
        uint4 aq, q0, q1, q2, q3;
        if (hasNext) {     // issue next tile's global loads before compute
            aq = loadA(kt + BK);
            const uint4* bq = (const uint4*)(Wt0s + (size_t)tid * K0pad + kt + BK);
            q0 = bq[0]; q1 = bq[1]; q2 = bq[2]; q3 = bq[3];
        }

        Frag a[2], b[4];
        #pragma unroll
        for (int mi = 0; mi < 2; ++mi) {
            int base = (waveM * 32 + mi * 16 + l) * SAS + half * 8;
            a[mi].q[0] = *(const uint4*)&sA[buf][base];
            a[mi].q[1] = *(const uint4*)&sA[buf][base + 16];
        }
        #pragma unroll
        for (int ni = 0; ni < 4; ++ni) {
            int base = (waveN * 64 + ni * 16 + l) * SBS + half * 8;
            b[ni].q[0] = *(const uint4*)&sB[buf][base];
            b[ni].q[1] = *(const uint4*)&sB[buf][base + 16];
        }
        #pragma unroll
        for (int mi = 0; mi < 2; ++mi)
            #pragma unroll
            for (int ni = 0; ni < 4; ++ni)
                acc[mi][ni] = __builtin_amdgcn_wmma_f32_16x16x32_bf16(
                    false, a[mi].v, false, b[ni].v, (short)0, acc[mi][ni], false, false);

        if (hasNext) {
            *(uint4*)&sA[buf ^ 1][arow * SAS + akg] = aq;
            uint4* sq = (uint4*)&sB[buf ^ 1][tid * SBS];
            sq[0] = q0; sq[1] = q1; sq[2] = q2; sq[3] = q3;
        }
    }
    epilogueToLds(b0 + (size_t)s * BN);    // hT not read by anyone yet: no barrier needed

    // ============ stages 2,3: h = silu(h x W{1,2} + b{1,2}), A from hT ============
    #pragma unroll 1
    for (int layer = 0; layer < 2; ++layer) {
        const uint16_t* Wl = layer ? Wt2s : Wt1s;
        const float*    bl = (layer ? b2 : b1) + (size_t)s * BN;
        zeroAcc();

        __syncthreads();   // hT writes visible to all; sB free for restaging
        {   // prologue: W tile k=0 into buffer 0
            const uint4* bq = (const uint4*)(Wl + (size_t)tid * BN);
            uint4 q0 = bq[0], q1 = bq[1], q2 = bq[2], q3 = bq[3];
            uint4* sq = (uint4*)&sB[0][tid * SBS];
            sq[0] = q0; sq[1] = q1; sq[2] = q2; sq[3] = q3;
        }
        int buf2 = 0;
        for (int kt = 0; kt < BN; kt += BK, buf2 ^= 1) {
            __syncthreads();
            const bool hasNext = (kt + BK) < BN;
            uint4 q0, q1, q2, q3;
            if (hasNext) {
                const uint4* bq = (const uint4*)(Wl + (size_t)tid * BN + kt + BK);
                q0 = bq[0]; q1 = bq[1]; q2 = bq[2]; q3 = bq[3];
            }

            Frag a[2], b[4];
            #pragma unroll
            for (int mi = 0; mi < 2; ++mi) {   // A fragments straight from resident hT
                int base = (waveM * 32 + mi * 16 + l) * HTS + kt + half * 8;
                a[mi].q[0] = *(const uint4*)&hT[base];
                a[mi].q[1] = *(const uint4*)&hT[base + 16];
            }
            #pragma unroll
            for (int ni = 0; ni < 4; ++ni) {
                int base = (waveN * 64 + ni * 16 + l) * SBS + half * 8;
                b[ni].q[0] = *(const uint4*)&sB[buf2][base];
                b[ni].q[1] = *(const uint4*)&sB[buf2][base + 16];
            }
            #pragma unroll
            for (int mi = 0; mi < 2; ++mi)
                #pragma unroll
                for (int ni = 0; ni < 4; ++ni)
                    acc[mi][ni] = __builtin_amdgcn_wmma_f32_16x16x32_bf16(
                        false, a[mi].v, false, b[ni].v, (short)0, acc[mi][ni], false, false);

            if (hasNext) {
                uint4* sq = (uint4*)&sB[buf2 ^ 1][tid * SBS];
                sq[0] = q0; sq[1] = q1; sq[2] = q2; sq[3] = q3;
            }
        }
        __syncthreads();        // all hT (A) reads done before overwriting with h(layer+1)
        epilogueToLds(bl);
    }

    // ================= stage 4: e = dot(h2, Wout) + bout; scatter-sum =================
    __syncthreads();            // h2 tile visible
    {
        const float* wp = Wout + (size_t)s * BN + lane * 8;
        float4 w0 = *(const float4*)wp;
        float4 w1 = *(const float4*)(wp + 4);
        const float bo = bout[s];
        const int* ids_s = ids + (size_t)s * Nrows;

        #pragma unroll
        for (int rr = 0; rr < 8; ++rr) {       // 8 rows per wave, 8 waves -> 64 rows
            const int row  = wave * 8 + rr;
            const int orow = m0 + row;
            union { uint4 q; uint16_t u[8]; } hv;
            hv.q = *(const uint4*)&hT[row * HTS + lane * 8];
            float sum = bf2f(hv.u[0]) * w0.x + bf2f(hv.u[1]) * w0.y +
                        bf2f(hv.u[2]) * w0.z + bf2f(hv.u[3]) * w0.w +
                        bf2f(hv.u[4]) * w1.x + bf2f(hv.u[5]) * w1.y +
                        bf2f(hv.u[6]) * w1.z + bf2f(hv.u[7]) * w1.w;
            #pragma unroll
            for (int off = 16; off > 0; off >>= 1)
                sum += __shfl_xor(sum, off, 32);
            if (lane == 0 && orow < Nrows)
                atomicAdd(&out[ids_s[orow]], sum + bo);
        }
    }
}

// ---------------- launch ----------------
extern "C" void kernel_launch(void* const* d_in, const int* in_sizes, int n_in,
                              void* d_out, int out_size, void* d_ws, size_t ws_size,
                              hipStream_t stream) {
    (void)in_sizes; (void)n_in; (void)ws_size; (void)out_size;
    constexpr int S = 4, N = 25000, DIN = 1296, KP0 = 1312, DH = 256, NSTRUCT = 1000;

    const float* feats = (const float*)d_in[0];
    const float* W0    = (const float*)d_in[1];
    const float* b0    = (const float*)d_in[2];
    const float* W1    = (const float*)d_in[3];
    const float* b1    = (const float*)d_in[4];
    const float* W2    = (const float*)d_in[5];
    const float* b2    = (const float*)d_in[6];
    const float* Wout  = (const float*)d_in[7];
    const float* bout  = (const float*)d_in[8];
    const int*   ids   = (const int*)d_in[9];
    float* out = (float*)d_out;

    // workspace: only converted/transposed bf16 weights (~3.7 MB)
    char* ws = (char*)d_ws;
    uint16_t* Wt0 = (uint16_t*)(ws + 0);          // 4*256*1312*2 = 2,686,976
    uint16_t* Wt1 = (uint16_t*)(ws + 2686976);    // 524,288
    uint16_t* Wt2 = (uint16_t*)(ws + 3211264);    // 524,288

    zero_out<<<4, 256, 0, stream>>>(out, NSTRUCT);

    convert_weight_T<<<dim3(DH, S), 256, 0, stream>>>(W0, Wt0, DIN, KP0);
    convert_weight_T<<<dim3(DH, S), 256, 0, stream>>>(W1, Wt1, DH, DH);
    convert_weight_T<<<dim3(DH, S), 256, 0, stream>>>(W2, Wt2, DH, DH);

    const int mtiles = (N + BM - 1) / BM;   // 391
    fused_mlp<<<dim3(mtiles, 1, S), 256, 0, stream>>>(
        feats, Wt0, Wt1, Wt2, b0, b1, b2, Wout, bout, ids, out, N, DIN, KP0);
}
